// CKAFormer_73100343378150
// MI455X (gfx1250) — matrix operand order repaired and checked by reading, not compiled, not run on hardware
//
#include <hip/hip_runtime.h>
#include <hip/hip_bf16.h>
#include <math.h>

// ---------------------------------------------------------------------------
// CKAFormer on MI455X (gfx1250, wave32, WMMA).
// N=32768, dim=512, hidden=16, out_dim=64, DEPTH=4, GAMMA=1e-4.
// Big GEMMs -> v_wmma_f32_16x16x32_bf16 (fp32 accumulate), register-tiled
// 32x64 per wave (2 A frags x 4 B frags = 8 WMMA per K-step) for fragment
// reuse. X / norms / MLP / softmax stay fp32. Transposed bf16 operand copies
// give contiguous per-lane K runs matching the ISA fragment layouts.
// ---------------------------------------------------------------------------

#define NROWS   32768
#define DIM     512
#define HID     16
#define ODIM    64
#define DEPTH   4
#define GAMMA   1e-4f

typedef __attribute__((ext_vector_type(16))) __bf16          v16bf;
typedef __attribute__((ext_vector_type(8)))  float           v8f;
typedef __attribute__((ext_vector_type(4)))  unsigned int    u32x4;
typedef __attribute__((ext_vector_type(4)))  float           f32x4;
typedef __attribute__((ext_vector_type(4)))  unsigned short  u16x4;

union Frag { u32x4 q[2]; v16bf v; };

__device__ __forceinline__ unsigned short f2bf(float f) {
    unsigned int u = __float_as_uint(f);
    u += 0x7FFFu + ((u >> 16) & 1u);        // round-to-nearest-even
    return (unsigned short)(u >> 16);
}

// A fragment, 16x32 bf16. Source row-major: rows = output M, contiguous K.
// Lane m=L&15 picks row; elems 0-7 = K[8g..8g+7], elems 8-15 = K[16+8g..23+8g].
__device__ __forceinline__ v16bf load_a(const unsigned short* __restrict__ base,
                                        int ld, int k0, int lane) {
    int m = lane & 15, g = lane >> 4;
    const unsigned short* p = base + (size_t)m * ld + k0 + 8 * g;
    Frag f;
    f.q[0] = *(const u32x4*)(p);
    f.q[1] = *(const u32x4*)(p + 16);
    return f.v;
}

// B fragment, 32x16 bf16, fed from a TRANSPOSED source Bt[col][k] (contig K).
// Lane col=L&15; elems 0-7 = K[8g..8g+7], elems 8-15 = K[16+8g..23+8g].
__device__ __forceinline__ v16bf load_bt(const unsigned short* __restrict__ base,
                                         int ld, int k0, int lane) {
    int c = lane & 15, g = lane >> 4;
    const unsigned short* p = base + (size_t)c * ld + k0 + 8 * g;
    Frag f;
    f.q[0] = *(const u32x4*)(p);
    f.q[1] = *(const u32x4*)(p + 16);
    return f.v;
}

__device__ __forceinline__ v8f wmma_bf16(v16bf a, v16bf b, v8f c) {
    return __builtin_amdgcn_wmma_f32_16x16x32_bf16(false, a, false, b,
                                                   (short)0, c, false, false);
}

// --------------------------- elementwise kernels ---------------------------

__global__ void k_zero_f32(float* __restrict__ p, int n) {
    int i = blockIdx.x * blockDim.x + threadIdx.x;
    if (i < n) p[i] = 0.0f;
}

__global__ void k_cvt_bf16(const float* __restrict__ s,
                           unsigned short* __restrict__ d, int n) {
    int i = blockIdx.x * blockDim.x + threadIdx.x;
    if (i < n) d[i] = f2bf(s[i]);
}

// Row-wise L2 normalize: X /= ||row||, emit bf16 copy. 1 block = 1 row.
__global__ void k_rownorm(float* __restrict__ X, unsigned short* __restrict__ Xb) {
    __shared__ float red[128];
    int n = blockIdx.x, t = threadIdx.x;           // 128 threads, 4 floats each
    f32x4 v = ((const f32x4*)(X + (size_t)n * DIM))[t];
    red[t] = v.x * v.x + v.y * v.y + v.z * v.z + v.w * v.w;
    __syncthreads();
    for (int o = 64; o > 0; o >>= 1) {
        if (t < o) red[t] += red[t + o];
        __syncthreads();
    }
    float inv = 1.0f / sqrtf(red[0]);
    v.x *= inv; v.y *= inv; v.z *= inv; v.w *= inv;
    ((f32x4*)(X + (size_t)n * DIM))[t] = v;
    u16x4 h; h.x = f2bf(v.x); h.y = f2bf(v.y); h.z = f2bf(v.z); h.w = f2bf(v.w);
    ((u16x4*)(Xb + (size_t)n * DIM))[t] = h;
}

// LDS-tiled bf16 transpose: dst[c][r] = src[r][c]. block (32,8).
__global__ void k_transpose16(const unsigned short* __restrict__ src,
                              unsigned short* __restrict__ dst, int R, int C) {
    __shared__ unsigned short tile[32][33];
    int c0 = blockIdx.x * 32, r0 = blockIdx.y * 32;
    int tx = threadIdx.x, ty = threadIdx.y;
    #pragma unroll
    for (int i = 0; i < 32; i += 8) {
        int r = r0 + ty + i, c = c0 + tx;
        if (r < R && c < C) tile[ty + i][tx] = src[(size_t)r * C + c];
    }
    __syncthreads();
    #pragma unroll
    for (int i = 0; i < 32; i += 8) {
        int r = c0 + ty + i, c = r0 + tx;
        if (r < C && c < R) dst[(size_t)r * R + c] = tile[tx][ty + i];
    }
}

// MLP (Linear 512->16, ReLU, Linear 16->64); mode 0: softmax -> Pb (bf16),
// mode 1: logits -> out (fp32). 1 block (64 threads) per row; tiny FLOPs.
__global__ void k_mlp(const float* __restrict__ X, const float* __restrict__ W1,
                      const float* __restrict__ b1, const float* __restrict__ W2,
                      const float* __restrict__ b2,
                      unsigned short* __restrict__ Pb, float* __restrict__ out,
                      int mode) {
    __shared__ float xrow[DIM];
    __shared__ float h[HID];
    __shared__ float lg[ODIM];
    int n = blockIdx.x, t = threadIdx.x;
    const float* xp = X + (size_t)n * DIM;
    for (int i = t; i < DIM; i += ODIM) xrow[i] = xp[i];
    __syncthreads();
    if (t < HID) {
        float a = b1[t];
        #pragma unroll 8
        for (int i = 0; i < DIM; ++i) a += xrow[i] * W1[i * HID + t];
        h[t] = fmaxf(a, 0.0f);
    }
    __syncthreads();
    float a = b2[t];
    #pragma unroll
    for (int j = 0; j < HID; ++j) a += h[j] * W2[j * ODIM + t];
    if (mode == 0) {
        lg[t] = a;
        __syncthreads();
        float mx = -INFINITY;
        for (int k = 0; k < ODIM; ++k) mx = fmaxf(mx, lg[k]);
        float s = 0.0f;
        for (int k = 0; k < ODIM; ++k) s += expf(lg[k] - mx);
        Pb[(size_t)n * ODIM + t] = f2bf(expf(a - mx) / s);
    } else {
        out[(size_t)n * ODIM + t] = a;
    }
}

// ------------------------------ WMMA GEMMs ---------------------------------

// PtXt[d][m] += sum_n P[n][m] * X[n][d]  ((P.T@X) stored transposed).
// M=16 (out_dim tile) x N=64 (1 A frag reused across 4 B frags); split-K over
// n with fp32 atomics. A from Pt[64][N], B from Xt[512][N].
__global__ void k_gemm_ptx(const unsigned short* __restrict__ Pt,
                           const unsigned short* __restrict__ Xt,
                           float* __restrict__ PtXt) {
    int lane = threadIdx.x;
    int m0 = blockIdx.x * 16, d0 = blockIdx.y * 64, n0 = blockIdx.z * 512;
    const unsigned short* a = Pt + (size_t)m0 * NROWS;
    v8f acc[4] = {};
    for (int k = 0; k < 512; k += 32) {
        v16bf av = load_a(a, NROWS, n0 + k, lane);
        #pragma unroll
        for (int j = 0; j < 4; ++j) {
            v16bf bv = load_bt(Xt + (size_t)(d0 + 16 * j) * NROWS, NROWS,
                               n0 + k, lane);
            acc[j] = wmma_bf16(av, bv, acc[j]);
        }
    }
    int g = lane >> 4, c = lane & 15;
    #pragma unroll
    for (int j = 0; j < 4; ++j) {
        float* o = PtXt + (size_t)(d0 + 16 * j + c) * ODIM + m0 + 8 * g;
        #pragma unroll
        for (int r = 0; r < 8; ++r) atomicAdd(&o[r], acc[j][r]);
    }
}

// X[n][d] += GAMMA * sum_k P[n][k] * PtX[k][d];  K=64. 32x64 tile per wave.
// Writes X (fp32) and refreshes Xb.
__global__ void k_gemm_pptx(const unsigned short* __restrict__ Pb,
                            const unsigned short* __restrict__ PtXtb,
                            float* __restrict__ X,
                            unsigned short* __restrict__ Xb) {
    int lane = threadIdx.x;
    int n0 = blockIdx.x * 32, d0 = blockIdx.y * 64;
    const unsigned short* a0 = Pb + (size_t)n0 * ODIM;
    const unsigned short* a1 = a0 + (size_t)16 * ODIM;
    v8f acc[2][4] = {};
    #pragma unroll
    for (int k = 0; k < ODIM; k += 32) {
        v16bf av0 = load_a(a0, ODIM, k, lane);
        v16bf av1 = load_a(a1, ODIM, k, lane);
        #pragma unroll
        for (int j = 0; j < 4; ++j) {
            v16bf bv = load_bt(PtXtb + (size_t)(d0 + 16 * j) * ODIM, ODIM, k, lane);
            acc[0][j] = wmma_bf16(av0, bv, acc[0][j]);
            acc[1][j] = wmma_bf16(av1, bv, acc[1][j]);
        }
    }
    int g = lane >> 4, c = lane & 15;
    #pragma unroll
    for (int i = 0; i < 2; ++i)
        #pragma unroll
        for (int j = 0; j < 4; ++j)
            #pragma unroll
            for (int r = 0; r < 8; ++r) {
                size_t idx = (size_t)(n0 + 16 * i + r + 8 * g) * DIM
                           + d0 + 16 * j + c;
                float xv = X[idx] + GAMMA * acc[i][j][r];
                X[idx] = xv;
                Xb[idx] = f2bf(xv);
            }
}

// C[di][dj] += sum_n X[n][di]*X[n][dj]; 32x64 tile, both operands from Xt,
// split-K over n with fp32 atomics.
__global__ void k_gemm_xtx(const unsigned short* __restrict__ Xt,
                           float* __restrict__ Cf) {
    int lane = threadIdx.x;
    int di0 = blockIdx.x * 32, dj0 = blockIdx.y * 64, n0 = blockIdx.z * 512;
    const unsigned short* a0 = Xt + (size_t)di0 * NROWS;
    const unsigned short* a1 = a0 + (size_t)16 * NROWS;
    v8f acc[2][4] = {};
    for (int k = 0; k < 512; k += 32) {
        v16bf av0 = load_a(a0, NROWS, n0 + k, lane);
        v16bf av1 = load_a(a1, NROWS, n0 + k, lane);
        #pragma unroll
        for (int j = 0; j < 4; ++j) {
            v16bf bv = load_bt(Xt + (size_t)(dj0 + 16 * j) * NROWS, NROWS,
                               n0 + k, lane);
            acc[0][j] = wmma_bf16(av0, bv, acc[0][j]);
            acc[1][j] = wmma_bf16(av1, bv, acc[1][j]);
        }
    }
    int g = lane >> 4, c = lane & 15;
    #pragma unroll
    for (int i = 0; i < 2; ++i)
        #pragma unroll
        for (int j = 0; j < 4; ++j) {
            float* o = Cf + (size_t)(di0 + 16 * i + 8 * g) * DIM + dj0 + 16 * j + c;
            #pragma unroll
            for (int r = 0; r < 8; ++r) atomicAdd(&o[(size_t)r * DIM], acc[i][j][r]);
        }
}

// X[n][d] -= GAMMA * sum_k X[n][k] * C[k][d]; 32x64 tile. C symmetric so
// row-major Cb serves as the transposed B source. Reads bf16 snapshot,
// writes fp32 only (race-free).
__global__ void k_gemm_xc(const unsigned short* __restrict__ Xb,
                          const unsigned short* __restrict__ Cb,
                          float* __restrict__ X) {
    int lane = threadIdx.x;
    int n0 = blockIdx.x * 32, d0 = blockIdx.y * 64;
    const unsigned short* a0 = Xb + (size_t)n0 * DIM;
    const unsigned short* a1 = a0 + (size_t)16 * DIM;
    v8f acc[2][4] = {};
    for (int k = 0; k < DIM; k += 32) {
        v16bf av0 = load_a(a0, DIM, k, lane);
        v16bf av1 = load_a(a1, DIM, k, lane);
        #pragma unroll
        for (int j = 0; j < 4; ++j) {
            v16bf bv = load_bt(Cb + (size_t)(d0 + 16 * j) * DIM, DIM, k, lane);
            acc[0][j] = wmma_bf16(av0, bv, acc[0][j]);
            acc[1][j] = wmma_bf16(av1, bv, acc[1][j]);
        }
    }
    int g = lane >> 4, c = lane & 15;
    #pragma unroll
    for (int i = 0; i < 2; ++i)
        #pragma unroll
        for (int j = 0; j < 4; ++j)
            #pragma unroll
            for (int r = 0; r < 8; ++r) {
                size_t idx = (size_t)(n0 + 16 * i + r + 8 * g) * DIM
                           + d0 + 16 * j + c;
                X[idx] -= GAMMA * acc[i][j][r];
            }
}

// ------------------------------ launcher -----------------------------------

extern "C" void kernel_launch(void* const* d_in, const int* in_sizes, int n_in,
                              void* d_out, int out_size, void* d_ws, size_t ws_size,
                              hipStream_t stream) {
    const float* Xin = (const float*)d_in[0];
    const float* W1  = (const float*)d_in[1];
    const float* b1  = (const float*)d_in[2];
    const float* W2  = (const float*)d_in[3];
    const float* b2  = (const float*)d_in[4];
    float* out = (float*)d_out;

    // Workspace carving (bytes): ~138 MB total.
    char* ws = (char*)d_ws;
    size_t off = 0;
    float*          Xf    = (float*)(ws + off);          off += (size_t)NROWS * DIM * 4;
    unsigned short* Xb    = (unsigned short*)(ws + off); off += (size_t)NROWS * DIM * 2;
    unsigned short* Xt    = (unsigned short*)(ws + off); off += (size_t)NROWS * DIM * 2;
    unsigned short* Pb    = (unsigned short*)(ws + off); off += (size_t)NROWS * ODIM * 2;
    unsigned short* Pt    = (unsigned short*)(ws + off); off += (size_t)NROWS * ODIM * 2;
    float*          PtXt  = (float*)(ws + off);          off += (size_t)DIM * ODIM * 4;
    unsigned short* PtXtb = (unsigned short*)(ws + off); off += (size_t)DIM * ODIM * 2;
    float*          Cf    = (float*)(ws + off);          off += (size_t)DIM * DIM * 4;
    unsigned short* Cb    = (unsigned short*)(ws + off); off += (size_t)DIM * DIM * 2;
    (void)off; (void)ws_size; (void)n_in; (void)in_sizes; (void)out_size;

    hipMemcpyAsync(Xf, Xin, (size_t)NROWS * DIM * 4, hipMemcpyDeviceToDevice, stream);

    dim3 tblk(32, 8, 1);
    for (int it = 0; it < DEPTH; ++it) {
        // 1) X = X / ||row||2 ; Xb = bf16(X)
        k_rownorm<<<NROWS, 128, 0, stream>>>(Xf, Xb);
        // 2) Xt = transpose(Xb)  [DIM][NROWS]
        k_transpose16<<<dim3(DIM / 32, NROWS / 32), tblk, 0, stream>>>(Xb, Xt, NROWS, DIM);
        // 3) P = softmax(MLP(X)) -> Pb bf16
        k_mlp<<<NROWS, ODIM, 0, stream>>>(Xf, W1, b1, W2, b2, Pb, nullptr, 0);
        // 4) Pt = transpose(Pb)  [ODIM][NROWS]
        k_transpose16<<<dim3(ODIM / 32, NROWS / 32), tblk, 0, stream>>>(Pb, Pt, NROWS, ODIM);
        // 5) PtXt = (P.T @ X).T  via split-K WMMA + fp32 atomics
        k_zero_f32<<<(DIM * ODIM + 255) / 256, 256, 0, stream>>>(PtXt, DIM * ODIM);
        k_gemm_ptx<<<dim3(ODIM / 16, DIM / 64, NROWS / 512), 32, 0, stream>>>(Pt, Xt, PtXt);
        k_cvt_bf16<<<(DIM * ODIM + 255) / 256, 256, 0, stream>>>(PtXt, PtXtb, DIM * ODIM);
        // 6) X += GAMMA * P @ PtX ; refresh Xb, then Xt
        k_gemm_pptx<<<dim3(NROWS / 32, DIM / 64), 32, 0, stream>>>(Pb, PtXtb, Xf, Xb);
        k_transpose16<<<dim3(DIM / 32, NROWS / 32), tblk, 0, stream>>>(Xb, Xt, NROWS, DIM);
        // 7) C = X.T @ X  via split-K WMMA + fp32 atomics
        k_zero_f32<<<(DIM * DIM + 255) / 256, 256, 0, stream>>>(Cf, DIM * DIM);
        k_gemm_xtx<<<dim3(DIM / 32, DIM / 64, NROWS / 512), 32, 0, stream>>>(Xt, Cf);
        k_cvt_bf16<<<(DIM * DIM + 255) / 256, 256, 0, stream>>>(Cf, Cb, DIM * DIM);
        // 8) X -= GAMMA * X @ C   (Xb is the pre-update snapshot: race-free)
        k_gemm_xc<<<dim3(NROWS / 32, DIM / 64), 32, 0, stream>>>(Xb, Cb, Xf);
    }
    // Final logits = MLP(X) -> d_out (fp32 [N,64])
    k_mlp<<<NROWS, ODIM, 0, stream>>>(Xf, W1, b1, W2, b2, nullptr, out, 1);
}
